// MultiHeadCoAttention_60017872994786
// MI455X (gfx1250) — matrix operand order repaired and checked by reading, not compiled, not run on hardware
//
#include <hip/hip_runtime.h>
#include <hip/hip_bf16.h>
#include <stdint.h>

#define B_  32
#define LT_ 512
#define LI_ 576
#define D_  768
#define H_  8
#define K_  512

// Scheduling fence: nothing moves across (keeps load groups ahead of WMMA groups
// so one relaxed s_wait_loadcnt covers a whole WMMA block).
#define SCHED_FENCE() __builtin_amdgcn_sched_barrier(0)

typedef __attribute__((ext_vector_type(8)))  __bf16 v8bf;
typedef __attribute__((ext_vector_type(16))) __bf16 v16bf;
typedef __attribute__((ext_vector_type(8)))  float  v8f;

union BFU { v16bf v; v8bf h[2]; };

static __device__ inline v8f wmma_bf16(v16bf a, v16bf b, v8f c) {
  // D = A(16x32) * B(32x16) + C, fp32 accumulate
  return __builtin_amdgcn_wmma_f32_16x16x32_bf16(false, a, false, b, (short)0, c, false, false);
}

static __device__ inline v8f vzero8() {
  v8f z = {0.f, 0.f, 0.f, 0.f, 0.f, 0.f, 0.f, 0.f};
  return z;
}

// A fragment: 16x32 bf16 tile, row-major with leading dim `ld` (global or LDS).
// Lane L holds row m = L%16, K chunks [kb, kb+8) and [16+kb, 16+kb+8), kb = (L<16?0:8).
static __device__ inline v16bf load_a_frag(const __bf16* tile, int ld) {
  int lane = threadIdx.x & 31;
  int m = lane & 15;
  int kb = (lane < 16) ? 0 : 8;
  const __bf16* p = tile + (size_t)m * ld + kb;
  BFU u;
  u.h[0] = *(const v8bf*)(p);
  u.h[1] = *(const v8bf*)(p + 16);
  return u.v;
}

// B fragment from an N x K (K-contiguous) matrix, leading dim `ld`.
// Lane L holds column n = L%16, K chunks [kb, kb+8) and [16+kb, 16+kb+8), kb = (L<16?0:8).
static __device__ inline v16bf load_b_frag(const __bf16* bt, int ld) {
  int lane = threadIdx.x & 31;
  int n = lane & 15;
  int ko = (lane < 16) ? 0 : 8;
  const __bf16* p = bt + (size_t)n * ld + ko;
  BFU u;
  u.h[0] = *(const v8bf*)(p);
  u.h[1] = *(const v8bf*)(p + 16);
  return u.v;
}

// ---------------------------------------------------------------------------
__global__ __launch_bounds__(256) void k_f32_to_bf16(const float* __restrict__ in,
                                                     __bf16* __restrict__ out, int n) {
  int i = blockIdx.x * blockDim.x + threadIdx.x;
  int stride = gridDim.x * blockDim.x;
  for (; i < n; i += stride) out[i] = (__bf16)in[i];
}

// ---------------------------------------------------------------------------
// Software-pipelined K loop with sched fences:
//   [loads for step s+1]  SCHED_FENCE  [T WMMAs for step s]  SCHED_FENCE
// P = A(M x E) @ Bt(N=512 x E)^T. grid (M/16); block 256; wave w owns n-subtiles w*4..+3.
template <int LROW>
__global__ __launch_bounds__(256) void k_proj(const __bf16* __restrict__ A,
                                              const __bf16* __restrict__ Bt,
                                              float* __restrict__ out,
                                              __bf16* __restrict__ outT) {
  constexpr int E = D_, N = K_, T = 4, STEPS = E / 32;
  int wave = threadIdx.x >> 5, lane = threadIdx.x & 31;
  int mt = blockIdx.x;
  const __bf16* Arow = A + (size_t)mt * 16 * E;
  const __bf16* Brow[T];
  for (int t = 0; t < T; ++t) Brow[t] = Bt + (size_t)((wave * T + t) * 16) * E;

  v8f acc[T];
  for (int t = 0; t < T; ++t) acc[t] = vzero8();
  v16bf a0 = load_a_frag(Arow, E);
  v16bf b0[T];
  for (int t = 0; t < T; ++t) b0[t] = load_b_frag(Brow[t], E);
#pragma unroll 2
  for (int s = 0; s < STEPS - 1; ++s) {
    int e1 = (s + 1) * 32;
    v16bf a1 = load_a_frag(Arow + e1, E);
    v16bf b1[T];
    for (int t = 0; t < T; ++t) b1[t] = load_b_frag(Brow[t] + e1, E);
    SCHED_FENCE();
    for (int t = 0; t < T; ++t) acc[t] = wmma_bf16(a0, b0[t], acc[t]);
    SCHED_FENCE();
    a0 = a1;
    for (int t = 0; t < T; ++t) b0[t] = b1[t];
  }
  for (int t = 0; t < T; ++t) acc[t] = wmma_bf16(a0, b0[t], acc[t]);

  for (int t = 0; t < T; ++t) {
    int n = (wave * T + t) * 16 + (lane & 15);
    for (int r = 0; r < 8; ++r) {
      int m = mt * 16 + ((lane < 16) ? r : r + 8);
      float v = acc[t][r];
      out[(size_t)m * N + n] = v;
      int b = m / LROW, x = m % LROW;
      outT[((size_t)b * N + n) * LROW + x] = (__bf16)v;
    }
  }
}

// ---------------------------------------------------------------------------
// Fused pt+aff: block = (xtile, b, h).
// Phase 1: pt_tile[16][768] = text_tile @ Wb[h]^T -> LDS (bf16); pipelined global frags.
// Phase 2: aff_tile[16][576] = pt_tile @ image[b]^T; A frags from LDS, B from global;
// one barrier between phases, none inside the K loops.
__global__ __launch_bounds__(256) void k_pt_aff(const __bf16* __restrict__ textb,
                                                const __bf16* __restrict__ imageb,
                                                const __bf16* __restrict__ Wbb,
                                                __bf16* __restrict__ aff,
                                                __bf16* __restrict__ affT) {
  __shared__ __align__(16) __bf16 sPT[16 * D_];  // 24 KB
  constexpr int STEPS = D_ / 32;
  int xt = blockIdx.x, b = blockIdx.y, h = blockIdx.z;
  int wave = threadIdx.x >> 5, lane = threadIdx.x & 31;
  const __bf16* Arow = textb + ((size_t)b * LT_ + (size_t)xt * 16) * D_;
  const __bf16* Wbh = Wbb + (size_t)h * D_ * D_;

  // Phase 1: N = 768 -> 48 subtiles, 6 per wave.
  {
    constexpr int T = 6;
    v8f acc[T];
    for (int t = 0; t < T; ++t) acc[t] = vzero8();
    v16bf a0 = load_a_frag(Arow, D_);
    v16bf b0[T];
    for (int t = 0; t < T; ++t) b0[t] = load_b_frag(Wbh + (size_t)((wave * T + t) * 16) * D_, D_);
#pragma unroll 2
    for (int s = 0; s < STEPS - 1; ++s) {
      int e1 = (s + 1) * 32;
      v16bf a1 = load_a_frag(Arow + e1, D_);
      v16bf b1[T];
      for (int t = 0; t < T; ++t)
        b1[t] = load_b_frag(Wbh + (size_t)((wave * T + t) * 16) * D_ + e1, D_);
      SCHED_FENCE();
      for (int t = 0; t < T; ++t) acc[t] = wmma_bf16(a0, b0[t], acc[t]);
      SCHED_FENCE();
      a0 = a1;
      for (int t = 0; t < T; ++t) b0[t] = b1[t];
    }
    for (int t = 0; t < T; ++t) acc[t] = wmma_bf16(a0, b0[t], acc[t]);
    for (int t = 0; t < T; ++t) {
      int n = (wave * T + t) * 16 + (lane & 15);
      for (int r = 0; r < 8; ++r) {
        int m = (lane < 16) ? r : r + 8;
        sPT[m * D_ + n] = (__bf16)acc[t][r];
      }
    }
  }
  __syncthreads();

  // Phase 2: N = 576 -> 36 subtiles; wave handles n-subtile = wave + 8*j (j < 5, ns < 36).
  {
    v8f acc2[5];
    for (int j = 0; j < 5; ++j) acc2[j] = vzero8();
    const __bf16* img = imageb + (size_t)b * LI_ * D_;
    v16bf a0 = load_a_frag(sPT, D_);
    v16bf b0[5];
    for (int j = 0; j < 5; ++j) {
      int ns = wave + 8 * j;
      if (ns < 36) b0[j] = load_b_frag(img + (size_t)ns * 16 * D_, D_);
    }
#pragma unroll 2
    for (int s = 0; s < STEPS - 1; ++s) {
      int e1 = (s + 1) * 32;
      v16bf a1 = load_a_frag(sPT + e1, D_);
      v16bf b1[5];
      for (int j = 0; j < 5; ++j) {
        int ns = wave + 8 * j;
        if (ns < 36) b1[j] = load_b_frag(img + (size_t)ns * 16 * D_ + e1, D_);
      }
      SCHED_FENCE();
      for (int j = 0; j < 5; ++j) {
        int ns = wave + 8 * j;
        if (ns < 36) acc2[j] = wmma_bf16(a0, b0[j], acc2[j]);
      }
      SCHED_FENCE();
      a0 = a1;
      for (int j = 0; j < 5; ++j) b0[j] = b1[j];
    }
    for (int j = 0; j < 5; ++j) {
      int ns = wave + 8 * j;
      if (ns < 36) acc2[j] = wmma_bf16(a0, b0[j], acc2[j]);
    }
    __bf16* affp  = aff  + ((size_t)h * B_ + b) * LT_ * LI_;
    __bf16* affTp = affT + ((size_t)h * B_ + b) * LI_ * LT_;
    for (int j = 0; j < 5; ++j) {
      int ns = wave + 8 * j;
      if (ns >= 36) break;
      int y = ns * 16 + (lane & 15);
      for (int r = 0; r < 8; ++r) {
        int x = xt * 16 + ((lane < 16) ? r : r + 8);
        __bf16 v = (__bf16)acc2[j][r];
        affp[(size_t)x * LI_ + y]  = v;
        affTp[(size_t)y * LT_ + x] = v;
      }
    }
  }
}

// ---------------------------------------------------------------------------
// Fused wqqc/wvvc + tanh + dot(wh) -> logits. A: [H,B,MROWS,RED] bf16; Bt: [B,K,RED] bf16;
// addm: [B,MROWS,K] fp32. grid (MROWS/16, B, H); block 256; wave w owns n-subtiles w+8t.
template <int MROWS, int RED>
__global__ __launch_bounds__(256) void k_logits(const __bf16* __restrict__ Amats,
                                                const __bf16* __restrict__ Bts,
                                                const float* __restrict__ addm,
                                                const float* __restrict__ wh,
                                                float* __restrict__ logits) {
  __shared__ float slog[16];
  constexpr int T = 4, STEPS = RED / 32;
  int mt = blockIdx.x, b = blockIdx.y, h = blockIdx.z;
  int wave = threadIdx.x >> 5, lane = threadIdx.x & 31;
  const __bf16* Arow = Amats + (((size_t)h * B_ + b) * MROWS + (size_t)mt * 16) * RED;
  const __bf16* Bt = Bts + (size_t)b * K_ * RED;
  const __bf16* Brow[T];
  for (int t = 0; t < T; ++t) Brow[t] = Bt + (size_t)((wave + 8 * t) * 16) * RED;

  v8f acc[T];
  for (int t = 0; t < T; ++t) acc[t] = vzero8();
  v16bf a0 = load_a_frag(Arow, RED);
  v16bf b0[T];
  for (int t = 0; t < T; ++t) b0[t] = load_b_frag(Brow[t], RED);
#pragma unroll 2
  for (int s = 0; s < STEPS - 1; ++s) {
    int e1 = (s + 1) * 32;
    v16bf a1 = load_a_frag(Arow + e1, RED);
    v16bf b1[T];
    for (int t = 0; t < T; ++t) b1[t] = load_b_frag(Brow[t] + e1, RED);
    SCHED_FENCE();
    for (int t = 0; t < T; ++t) acc[t] = wmma_bf16(a0, b0[t], acc[t]);
    SCHED_FENCE();
    a0 = a1;
    for (int t = 0; t < T; ++t) b0[t] = b1[t];
  }
  for (int t = 0; t < T; ++t) acc[t] = wmma_bf16(a0, b0[t], acc[t]);

  if (threadIdx.x < 16) slog[threadIdx.x] = 0.f;
  __syncthreads();
  float part[8];
  for (int r = 0; r < 8; ++r) part[r] = 0.f;
  for (int t = 0; t < T; ++t) {
    int k = (wave + 8 * t) * 16 + (lane & 15);
    float whk = wh[k];
    for (int r = 0; r < 8; ++r) {
      int mrow = mt * 16 + ((lane < 16) ? r : r + 8);
      float v = tanhf(acc[t][r] + addm[((size_t)b * MROWS + mrow) * K_ + k]);
      part[r] += v * whk;
    }
  }
  for (int r = 0; r < 8; ++r) {
    int ml = (lane < 16) ? r : r + 8;
    atomicAdd(&slog[ml], part[r]);
  }
  __syncthreads();
  if (threadIdx.x < 16)
    logits[((size_t)h * B_ + b) * MROWS + mt * 16 + threadIdx.x] = slog[threadIdx.x];
}

// ---------------------------------------------------------------------------
// Softmax over logits + context accumulation + combine. grid (B, H); block 256.
__global__ __launch_bounds__(256) void k_ctx(const float* __restrict__ logv,
                                             const float* __restrict__ logq,
                                             const float* __restrict__ image,
                                             const float* __restrict__ text,
                                             __bf16* __restrict__ comb) {
  __shared__ float sm[LI_ + LT_];
  __shared__ float rtmp[256];
  int b = blockIdx.x, h = blockIdx.y, tid = threadIdx.x;
  const float* lv = logv + ((size_t)h * B_ + b) * LI_;
  const float* lq = logq + ((size_t)h * B_ + b) * LT_;
  for (int i = tid; i < LI_; i += 256) sm[i] = lv[i];
  for (int i = tid; i < LT_; i += 256) sm[LI_ + i] = lq[i];
  __syncthreads();
  for (int seg = 0; seg < 2; ++seg) {
    int base = seg ? LI_ : 0;
    int len = seg ? LT_ : LI_;
    float mv = -3.0e38f;
    for (int i = tid; i < len; i += 256) mv = fmaxf(mv, sm[base + i]);
    rtmp[tid] = mv; __syncthreads();
    for (int s = 128; s > 0; s >>= 1) {
      if (tid < s) rtmp[tid] = fmaxf(rtmp[tid], rtmp[tid + s]);
      __syncthreads();
    }
    mv = rtmp[0]; __syncthreads();
    float ps = 0.f;
    for (int i = tid; i < len; i += 256) {
      float e = __expf(sm[base + i] - mv);
      sm[base + i] = e;
      ps += e;
    }
    rtmp[tid] = ps; __syncthreads();
    for (int s = 128; s > 0; s >>= 1) {
      if (tid < s) rtmp[tid] += rtmp[tid + s];
      __syncthreads();
    }
    float inv = 1.f / rtmp[0]; __syncthreads();
    for (int i = tid; i < len; i += 256) sm[base + i] *= inv;
    __syncthreads();
  }
  const float* img = image + (size_t)b * LI_ * D_;
  const float* txt = text + (size_t)b * LT_ * D_;
  for (int d = tid; d < D_; d += 256) {
    float acc = 0.f;
    for (int y = 0; y < LI_; ++y) acc += sm[y] * img[(size_t)y * D_ + d];
    for (int x = 0; x < LT_; ++x) acc += sm[LI_ + x] * txt[(size_t)x * D_ + d];
    comb[((size_t)b * H_ + h) * D_ + d] = (__bf16)acc;
  }
}

// ---------------------------------------------------------------------------
// out[32,768] = combined(32 x 6144) @ Wo(768 x 6144)^T + bo. grid (2); block 256;
// wave w owns n-subtiles w*6 .. w*6+5 (N=768 -> 48 subtiles). Pipelined + fences.
__global__ __launch_bounds__(256) void k_out(const __bf16* __restrict__ comb,
                                             const __bf16* __restrict__ Wob,
                                             const float* __restrict__ bo,
                                             float* __restrict__ out) {
  constexpr int E = H_ * D_, T = 6, STEPS = E / 32;
  int mt = blockIdx.x;
  int wave = threadIdx.x >> 5, lane = threadIdx.x & 31;
  const __bf16* Arow = comb + (size_t)mt * 16 * E;
  const __bf16* Brow[T];
  for (int t = 0; t < T; ++t) Brow[t] = Wob + (size_t)((wave * T + t) * 16) * E;

  v8f acc[T];
  for (int t = 0; t < T; ++t) acc[t] = vzero8();
  v16bf a0 = load_a_frag(Arow, E);
  v16bf b0[T];
  for (int t = 0; t < T; ++t) b0[t] = load_b_frag(Brow[t], E);
#pragma unroll 2
  for (int s = 0; s < STEPS - 1; ++s) {
    int e1 = (s + 1) * 32;
    v16bf a1 = load_a_frag(Arow + e1, E);
    v16bf b1[T];
    for (int t = 0; t < T; ++t) b1[t] = load_b_frag(Brow[t] + e1, E);
    SCHED_FENCE();
    for (int t = 0; t < T; ++t) acc[t] = wmma_bf16(a0, b0[t], acc[t]);
    SCHED_FENCE();
    a0 = a1;
    for (int t = 0; t < T; ++t) b0[t] = b1[t];
  }
  for (int t = 0; t < T; ++t) acc[t] = wmma_bf16(a0, b0[t], acc[t]);

  for (int t = 0; t < T; ++t) {
    int n = (wave * T + t) * 16 + (lane & 15);
    for (int r = 0; r < 8; ++r) {
      int m = mt * 16 + ((lane < 16) ? r : r + 8);
      out[(size_t)m * D_ + n] = acc[t][r] + bo[n];
    }
  }
}

// ---------------------------------------------------------------------------
extern "C" void kernel_launch(void* const* d_in, const int* in_sizes, int n_in,
                              void* d_out, int out_size, void* d_ws, size_t ws_size,
                              hipStream_t stream) {
  (void)in_sizes; (void)n_in; (void)out_size; (void)ws_size;
  const float* text  = (const float*)d_in[0];
  const float* image = (const float*)d_in[1];
  // d_in[2] = text_mask: unused by the reference math (all-ones)
  const float* Wb  = (const float*)d_in[3];
  const float* Wv  = (const float*)d_in[4];
  const float* Wq  = (const float*)d_in[5];
  const float* Whv = (const float*)d_in[6];
  const float* Whq = (const float*)d_in[7];
  const float* Wo  = (const float*)d_in[8];
  const float* bo  = (const float*)d_in[9];
  float* out = (float*)d_out;

  char* ws = (char*)d_ws;
  size_t off = 0;
  auto alloc = [&](size_t bytes) -> char* {
    char* p = ws + off;
    off += (bytes + 255) & ~(size_t)255;
    return p;
  };
  __bf16* textb  = (__bf16*)alloc((size_t)B_ * LT_ * D_ * 2);
  __bf16* imageb = (__bf16*)alloc((size_t)B_ * LI_ * D_ * 2);
  __bf16* Wbb    = (__bf16*)alloc((size_t)H_ * D_ * D_ * 2);
  __bf16* Wvb    = (__bf16*)alloc((size_t)K_ * D_ * 2);
  __bf16* Wqb    = (__bf16*)alloc((size_t)K_ * D_ * 2);
  __bf16* Wob    = (__bf16*)alloc((size_t)D_ * H_ * D_ * 2);
  float*  wqq    = (float*) alloc((size_t)B_ * LT_ * K_ * 4);
  __bf16* wqqT   = (__bf16*)alloc((size_t)B_ * K_ * LT_ * 2);
  float*  wvv    = (float*) alloc((size_t)B_ * LI_ * K_ * 4);
  __bf16* wvvT   = (__bf16*)alloc((size_t)B_ * K_ * LI_ * 2);
  __bf16* aff    = (__bf16*)alloc((size_t)H_ * B_ * LT_ * LI_ * 2);
  __bf16* affT   = (__bf16*)alloc((size_t)H_ * B_ * LI_ * LT_ * 2);
  float*  logv   = (float*) alloc((size_t)H_ * B_ * LI_ * 4);
  float*  logq   = (float*) alloc((size_t)H_ * B_ * LT_ * 4);
  __bf16* comb   = (__bf16*)alloc((size_t)B_ * H_ * D_ * 2);

  // 1) fp32 -> bf16 conversions
  k_f32_to_bf16<<<2048, 256, 0, stream>>>(text,  textb,  B_ * LT_ * D_);
  k_f32_to_bf16<<<2048, 256, 0, stream>>>(image, imageb, B_ * LI_ * D_);
  k_f32_to_bf16<<<2048, 256, 0, stream>>>(Wb, Wbb, H_ * D_ * D_);
  k_f32_to_bf16<<<512,  256, 0, stream>>>(Wv, Wvb, K_ * D_);
  k_f32_to_bf16<<<512,  256, 0, stream>>>(Wq, Wqb, K_ * D_);
  k_f32_to_bf16<<<2048, 256, 0, stream>>>(Wo, Wob, D_ * H_ * D_);

  // 2) wq_q = text @ Wq^T ; wv_v = image @ Wv^T (fp32 + bf16 K-major copies)
  k_proj<LT_><<<dim3(B_ * LT_ / 16), 256, 0, stream>>>(textb,  Wqb, wqq, wqqT);
  k_proj<LI_><<<dim3(B_ * LI_ / 16), 256, 0, stream>>>(imageb, Wvb, wvv, wvvT);

  // 3) fused pt + aff
  k_pt_aff<<<dim3(LT_ / 16, B_, H_), 256, 0, stream>>>(textb, imageb, Wbb, aff, affT);

  // 4) fused wqqc/wvvc + tanh + dot -> logits
  k_logits<LI_, LT_><<<dim3(LI_ / 16, B_, H_), 256, 0, stream>>>(affT, wqqT, wvv, Whv, logv);
  k_logits<LT_, LI_><<<dim3(LT_ / 16, B_, H_), 256, 0, stream>>>(aff,  wvvT, wqq, Whq, logq);

  // 5) softmax + context + combine
  k_ctx<<<dim3(B_, H_), 256, 0, stream>>>(logv, logq, image, text, comb);

  // 6) output projection
  k_out<<<dim3(2), 256, 0, stream>>>(comb, Wob, bo, out);
}
